// Kernel_65481071400600
// MI455X (gfx1250) — matrix operand order repaired and checked by reading, not compiled, not run on hardware
//
#include <hip/hip_runtime.h>

typedef __attribute__((ext_vector_type(16))) __bf16 v16bf;
typedef __attribute__((ext_vector_type(8)))  __bf16 v8bf;
typedef __attribute__((ext_vector_type(8)))  float  v8f;
typedef __attribute__((ext_vector_type(8)))  unsigned short ushort8v;
typedef __attribute__((ext_vector_type(4)))  unsigned short ushort4v;
typedef __attribute__((ext_vector_type(4)))  int i32x4;

#define NROWS 8192
#define KDIM  256
#define BM    128
#define BN    64
#define BK    32
#define LDK   40            // padded LDS K-stride: 80B rows -> conflict-free b128 frag loads
#define NKT   (KDIM / BK)

#define AS1 __attribute__((address_space(1)))
#define AS3 __attribute__((address_space(3)))

#ifndef __has_builtin
#define __has_builtin(x) 0
#endif
#if defined(__gfx1250__) && __has_builtin(__builtin_amdgcn_global_load_async_to_lds_b128)
#define HAVE_ASYNC 1
#else
#define HAVE_ASYNC 0
#endif

__device__ __forceinline__ unsigned short f32_to_bf16_rne(float f) {
    unsigned u = __builtin_bit_cast(unsigned, f);
    u += 0x7FFFu + ((u >> 16) & 1u);
    return (unsigned short)(u >> 16);
}
__device__ __forceinline__ float bf16u_to_f32(unsigned short h) {
    unsigned u = ((unsigned)h) << 16;
    return __builtin_bit_cast(float, u);
}
__device__ __forceinline__ v16bf frag_combine(v8bf a, v8bf b) {
    v16bf r;
#pragma unroll
    for (int i = 0; i < 8; ++i) { r[i] = a[i]; r[8 + i] = b[i]; }
    return r;
}

#if HAVE_ASYNC
__device__ __forceinline__ void async_ld_b128(const void* gp, void* lp) {
    __builtin_amdgcn_global_load_async_to_lds_b128(
        (AS1 i32x4*)(uintptr_t)gp,
        (AS3 i32x4*)(AS3 void*)lp, 0, 0);
}
#endif

// ---- one-shot split: a(f32) -> hi = bf16(a), lo = bf16(a - hi) ----------
__global__ __launch_bounds__(256) void split_bf16x2(const float* __restrict__ a,
                                                    unsigned short* __restrict__ hi,
                                                    unsigned short* __restrict__ lo) {
    int i = (blockIdx.x * 256 + threadIdx.x) * 4;
    float4 v = *(const float4*)(a + i);
    float f[4] = {v.x, v.y, v.z, v.w};
    ushort4v h, l;
#pragma unroll
    for (int j = 0; j < 4; ++j) {
        unsigned short hj = f32_to_bf16_rne(f[j]);
        h[j] = hj;
        l[j] = f32_to_bf16_rne(f[j] - bf16u_to_f32(hj));
    }
    *(ushort4v*)(hi + i) = h;
    *(ushort4v*)(lo + i) = l;
}

// ---- row-wise sum of squares (f32 source, full precision) ---------------
__global__ __launch_bounds__(256) void row_sumsq(const float* __restrict__ a,
                                                 float* __restrict__ out) {
    int r = blockIdx.x * 256 + threadIdx.x;
    const float4* p = (const float4*)(a + (size_t)r * KDIM);
    float s = 0.f;
#pragma unroll 8
    for (int i = 0; i < KDIM / 4; ++i) {
        float4 v = p[i];
        s += v.x * v.x + v.y * v.y + v.z * v.z + v.w * v.w;
    }
    out[r] = s;
}

// ---- fused RBF: out[m][n] = exp(-(x2[m]+y2[n]-2*x_m.y_n)) ---------------
// bf16x3 split-precision GEMM on V_WMMA_F32_16X16X32_BF16, async-LDS staged.
__global__ __launch_bounds__(256) void rbf_wmma(const unsigned short* __restrict__ xhi,
                                                const unsigned short* __restrict__ xlo,
                                                const unsigned short* __restrict__ yhi,
                                                const unsigned short* __restrict__ ylo,
                                                const float* __restrict__ x2,
                                                const float* __restrict__ y2,
                                                float* __restrict__ out) {
    __shared__ unsigned short sXhi[2][BM * LDK];
    __shared__ unsigned short sXlo[2][BM * LDK];
    __shared__ unsigned short sYhi[2][BN * LDK];
    __shared__ unsigned short sYlo[2][BN * LDK];

    const int tid  = threadIdx.x;
    const int lane = tid & 31;
    const int wave = tid >> 5;      // 0..7
    const int wm   = wave & 3;      // 4 waves along M
    const int wn   = wave >> 2;     // 2 waves along N
    const int m0   = blockIdx.y * BM;
    const int n0   = blockIdx.x * BN;
    const int h    = lane >> 4;     // half-wave select
    const int l16  = lane & 15;

    // stage one K-step tile (pre-split bf16) into LDS buffer b
    auto stage = [&](int b, int k0) {
#if HAVE_ASYNC
        // 6 async b128 issues per wave, per-lane global->LDS, ASYNCcnt-tracked
#pragma unroll
        for (int i = 0; i < 2; ++i) {               // X: 128 rows x 4 chunks
            int li = i * 256 + tid;
            int row = li >> 2, c = (li & 3) * 8;
            size_t g = (size_t)(m0 + row) * KDIM + k0 + c;
            int lo_off = row * LDK + c;
            async_ld_b128(xhi + g, &sXhi[b][lo_off]);
            async_ld_b128(xlo + g, &sXlo[b][lo_off]);
        }
        {                                           // Y: 64 rows x 4 chunks
            int row = tid >> 2, c = (tid & 3) * 8;
            size_t g = (size_t)(n0 + row) * KDIM + k0 + c;
            int lo_off = row * LDK + c;
            async_ld_b128(yhi + g, &sYhi[b][lo_off]);
            async_ld_b128(ylo + g, &sYlo[b][lo_off]);
        }
#else
        // fallback: plain b128 loads + ds stores (no conversion in hot loop)
#pragma unroll
        for (int i = 0; i < 2; ++i) {
            int li = i * 256 + tid;
            int row = li >> 2, c = (li & 3) * 8;
            size_t g = (size_t)(m0 + row) * KDIM + k0 + c;
            int lo_off = row * LDK + c;
            *(ushort8v*)&sXhi[b][lo_off] = *(const ushort8v*)(xhi + g);
            *(ushort8v*)&sXlo[b][lo_off] = *(const ushort8v*)(xlo + g);
        }
        {
            int row = tid >> 2, c = (tid & 3) * 8;
            size_t g = (size_t)(n0 + row) * KDIM + k0 + c;
            int lo_off = row * LDK + c;
            *(ushort8v*)&sYhi[b][lo_off] = *(const ushort8v*)(yhi + g);
            *(ushort8v*)&sYlo[b][lo_off] = *(const ushort8v*)(ylo + g);
        }
#endif
    };

    v8f acc[2][2];
#pragma unroll
    for (int mi = 0; mi < 2; ++mi)
#pragma unroll
        for (int ni = 0; ni < 2; ++ni)
            acc[mi][ni] = {};

#if HAVE_ASYNC
    stage(0, 0);                    // prologue: prefetch first tile
#endif

    for (int kt = 0; kt < NKT; ++kt) {
        const int b = HAVE_ASYNC ? (kt & 1) : 0;

        __syncthreads();            // all waves done reading buffer being refilled below
#if HAVE_ASYNC
        if (kt + 1 < NKT) {
            stage((kt + 1) & 1, (kt + 1) * BK);     // 6 new asyncs -> cnt <= 12
            asm volatile("s_wait_asynccnt 0x6" ::: "memory");  // current tile landed
        } else {
            asm volatile("s_wait_asynccnt 0x0" ::: "memory");
        }
#else
        stage(0, kt * BK);
#endif
        __syncthreads();            // everyone's share of current tile landed

        const unsigned short* pXhi = sXhi[b];
        const unsigned short* pXlo = sXlo[b];
        const unsigned short* pYhi = sYhi[b];
        const unsigned short* pYlo = sYlo[b];

        // A 16x32 bf16 frag: lane half h -> K chunks {h*8..+7, h*8+16..+23}
        v16bf ahi[2], alo[2];
#pragma unroll
        for (int mi = 0; mi < 2; ++mi) {
            int row = (wm * 32 + mi * 16 + l16) * LDK;
            ahi[mi] = frag_combine(*(const v8bf*)&pXhi[row + h * 8],
                                   *(const v8bf*)&pXhi[row + h * 8 + 16]);
            alo[mi] = frag_combine(*(const v8bf*)&pXlo[row + h * 8],
                                   *(const v8bf*)&pXlo[row + h * 8 + 16]);
        }
        // B 32x16 bf16 frag: column = lane&15 (= y row, B = y^T), K chunks {h*8..+7, h*8+16..+23}
        v16bf bhi[2], blo[2];
#pragma unroll
        for (int ni = 0; ni < 2; ++ni) {
            int row = (wn * 32 + ni * 16 + l16) * LDK;
            bhi[ni] = frag_combine(*(const v8bf*)&pYhi[row + h * 8],
                                   *(const v8bf*)&pYhi[row + h * 8 + 16]);
            blo[ni] = frag_combine(*(const v8bf*)&pYlo[row + h * 8],
                                   *(const v8bf*)&pYlo[row + h * 8 + 16]);
        }

        // bf16x3 split-precision accumulate: hi*hi + hi*lo + lo*hi (f32 acc)
#pragma unroll
        for (int mi = 0; mi < 2; ++mi)
#pragma unroll
            for (int ni = 0; ni < 2; ++ni) {
                acc[mi][ni] = __builtin_amdgcn_wmma_f32_16x16x32_bf16(
                    false, ahi[mi], false, bhi[ni], (short)0, acc[mi][ni], false, false);
                acc[mi][ni] = __builtin_amdgcn_wmma_f32_16x16x32_bf16(
                    false, ahi[mi], false, blo[ni], (short)0, acc[mi][ni], false, false);
                acc[mi][ni] = __builtin_amdgcn_wmma_f32_16x16x32_bf16(
                    false, alo[mi], false, bhi[ni], (short)0, acc[mi][ni], false, false);
            }
    }

    // ---- fused RBF epilogue: C/D layout M = r + 8*h, N = lane&15 ----
#pragma unroll
    for (int mi = 0; mi < 2; ++mi)
#pragma unroll
        for (int ni = 0; ni < 2; ++ni) {
            int n = n0 + wn * 32 + ni * 16 + l16;
            float ysq = y2[n];
#pragma unroll
            for (int r = 0; r < 8; ++r) {
                int m = m0 + wm * 32 + mi * 16 + r + 8 * h;
                float d2 = x2[m] + ysq - 2.0f * acc[mi][ni][r];
                d2 = fmaxf(d2, 0.0f);
                out[(size_t)m * NROWS + n] = __expf(-d2);   // GAMMA = 1.0
            }
        }
}

extern "C" void kernel_launch(void* const* d_in, const int* in_sizes, int n_in,
                              void* d_out, int out_size, void* d_ws, size_t ws_size,
                              hipStream_t stream) {
    const float* x = (const float*)d_in[0];
    const float* y = (const float*)d_in[1];
    float* out = (float*)d_out;

    // d_ws layout: 4x bf16[8192][256] (16 MB) + x2/y2 norms (64 KB)
    unsigned short* xhi = (unsigned short*)d_ws;
    unsigned short* xlo = xhi + (size_t)NROWS * KDIM;
    unsigned short* yhi = xlo + (size_t)NROWS * KDIM;
    unsigned short* ylo = yhi + (size_t)NROWS * KDIM;
    float* x2 = (float*)(ylo + (size_t)NROWS * KDIM);
    float* y2 = x2 + NROWS;

    const int nsplit = (NROWS * KDIM / 4) / 256;    // 2048 blocks
    split_bf16x2<<<nsplit, 256, 0, stream>>>(x, xhi, xlo);
    split_bf16x2<<<nsplit, 256, 0, stream>>>(y, yhi, ylo);
    row_sumsq<<<NROWS / 256, 256, 0, stream>>>(x, x2);
    row_sumsq<<<NROWS / 256, 256, 0, stream>>>(y, y2);

    dim3 grid(NROWS / BN, NROWS / BM);              // 128 x 64 blocks
    rbf_wmma<<<grid, 256, 0, stream>>>(xhi, xlo, yhi, ylo, x2, y2, out);
}
